// Encoder_79310866088428
// MI455X (gfx1250) — matrix operand order repaired and checked by reading, not compiled, not run on hardware
//
#include <hip/hip_runtime.h>
#include <hip/hip_bf16.h>
#include <math.h>

// ---------------------------------------------------------------------------
// CDNA5 / gfx1250 implementation. wave32. Matrix math via v_wmma_f32_16x16x32_f16.
// B-panel staged through LDS with global_load_async_to_lds_b128 (+ s_wait_asynccnt),
// A streamed from global as b128.
// ---------------------------------------------------------------------------

typedef __attribute__((ext_vector_type(16))) _Float16 v16h;
typedef __attribute__((ext_vector_type(8)))  float    v8f;

#define CH 256   // IN_CH
#define BK 128   // K-panel staged in LDS per step (must divide all K used)

#if defined(__AMDGCN__) && __has_builtin(__builtin_amdgcn_global_load_async_to_lds_b128) && __has_builtin(__builtin_amdgcn_s_wait_asynccnt)
#define USE_ASYNC_LDS 1
// exact parameter types per the clang diagnostic: vector_size(16) int, AS1 / AS3
typedef int v4i_b128 __attribute__((vector_size(16)));
typedef __attribute__((address_space(1))) v4i_b128* g_v4i_p;
typedef __attribute__((address_space(3))) v4i_b128* l_v4i_p;
#else
#define USE_ASYNC_LDS 0
#endif

// per-lane epilogue shared by both GEMM kernels
__device__ __forceinline__ void gemm_epilogue(const v8f& acc, float* __restrict__ Y,
                                              int ldy, int m0, int n0, int mr, int hi,
                                              const float* __restrict__ bias,
                                              float alpha, int accum, int act)
{
#pragma unroll
    for (int r = 0; r < 8; ++r) {
        int m = m0 + hi * 8 + r;
        int n = n0 + mr;
        float v = alpha * acc[r];
        if (bias)  v += bias[n];
        if (accum) v += Y[(size_t)m * ldy + n];
        if (act == 1)      v = 1.f / (1.f + __expf(-v));   // sigmoid
        else if (act == 2) v = tanhf(v);                    // tanh
        Y[(size_t)m * ldy + n] = v;
    }
}

// load the 16x32 A tile for this lane as f16 (contiguous k order):
//   k = hi*16 + e
__device__ __forceinline__ v16h load_a_tile(const float* __restrict__ xrow, int kbase, int hi)
{
    const float4 a0 = *reinterpret_cast<const float4*>(xrow + kbase + hi * 16);
    const float4 a1 = *reinterpret_cast<const float4*>(xrow + kbase + hi * 16 + 4);
    const float4 a2 = *reinterpret_cast<const float4*>(xrow + kbase + hi * 16 + 8);
    const float4 a3 = *reinterpret_cast<const float4*>(xrow + kbase + hi * 16 + 12);
    v16h a;
    a[0] = (_Float16)a0.x;  a[1] = (_Float16)a0.y;  a[2]  = (_Float16)a0.z;  a[3]  = (_Float16)a0.w;
    a[4] = (_Float16)a1.x;  a[5] = (_Float16)a1.y;  a[6]  = (_Float16)a1.z;  a[7]  = (_Float16)a1.w;
    a[8] = (_Float16)a2.x;  a[9] = (_Float16)a2.y;  a[10] = (_Float16)a2.z;  a[11] = (_Float16)a2.w;
    a[12] = (_Float16)a3.x; a[13] = (_Float16)a3.y; a[14] = (_Float16)a3.z;  a[15] = (_Float16)a3.w;
    return a;
}

// --------------------------- GEMM (W normal) -------------------------------
// Y[R x Cc] = act( alpha * X[R x K] @ W[K x Cc] + bias (+Y) )
// Requires: R % 128 == 0, Cc % 16 == 0, K % 128 == 0 (true for all call sites).
// All 8 waves of a block share the same 16-column W panel -> stage it in LDS.
__global__ __launch_bounds__(256)
void gemm_nn(const float* __restrict__ X, int ldx,
             const float* __restrict__ W, int ldw,
             const float* __restrict__ bias,
             float* __restrict__ Y, int ldy,
             int R, int K, int Cc,
             float alpha, int accum, int act)
{
    __shared__ float Bs[BK * 16];

    const int lane = threadIdx.x & 31;
    const int wave = threadIdx.x >> 5;
    const int n0   = blockIdx.x * 16;
    const int m0   = (blockIdx.y * 8 + wave) * 16;
    const int mr   = lane & 15;
    const int hi   = lane >> 4;

    // staging mapping: thread t copies 8 consecutive floats of one W row
    const int tkr   = threadIdx.x >> 1;         // 0..127 : k-row within panel
    const int thalf = (threadIdx.x & 1) * 8;    // 0 or 8 : column offset

    v8f acc = {0.f, 0.f, 0.f, 0.f, 0.f, 0.f, 0.f, 0.f};
    const float* xrow = X + (size_t)(m0 + mr) * ldx;

    for (int k0 = 0; k0 < K; k0 += BK) {
        const float* gsrc = W + (size_t)(k0 + tkr) * ldw + n0 + thalf;
        float*       ldst = &Bs[tkr * 16 + thalf];
#if USE_ASYNC_LDS
        __builtin_amdgcn_global_load_async_to_lds_b128((g_v4i_p)gsrc, (l_v4i_p)ldst, 0, 0);
        __builtin_amdgcn_global_load_async_to_lds_b128((g_v4i_p)gsrc, (l_v4i_p)ldst, 16, 0);
        __builtin_amdgcn_s_wait_asynccnt(0);
#else
        *reinterpret_cast<float4*>(ldst)     = *reinterpret_cast<const float4*>(gsrc);
        *reinterpret_cast<float4*>(ldst + 4) = *reinterpret_cast<const float4*>(gsrc + 4);
#endif
        __syncthreads();

#pragma unroll
        for (int kc = 0; kc < BK; kc += 32) {
            __builtin_prefetch(xrow + k0 + kc + 256, 0, 1);      // global_prefetch_b8
            v16h a = load_a_tile(xrow, k0 + kc, hi);
            v16h b;
#pragma unroll
            for (int e = 0; e < 16; ++e)                          // B: k = hi*16 + e, col mr
                b[e] = (_Float16)Bs[(kc + hi * 16 + e) * 16 + mr];
            acc = __builtin_amdgcn_wmma_f32_16x16x32_f16(false, a, false, b,
                                                         (short)0, acc, false, false);
        }
        __syncthreads();
    }
    gemm_epilogue(acc, Y, ldy, m0, n0, mr, hi, bias, alpha, accum, act);
}

// ------------------------- GEMM (W transposed) -----------------------------
// Y[R x Cc] = alpha * X[R x K] @ W^T, W stored [Cc x K] row-major (ldw = K).
// Used for S = q @ k^T. Per-lane B loads are 16 consecutive floats -> b128.
__global__ __launch_bounds__(256)
void gemm_nt(const float* __restrict__ X, int ldx,
             const float* __restrict__ W, int ldw,
             float* __restrict__ Y, int ldy,
             int R, int K, int Cc, float alpha)
{
    const int lane = threadIdx.x & 31;
    const int wave = threadIdx.x >> 5;
    const int n0   = blockIdx.x * 16;
    const int m0   = (blockIdx.y * 8 + wave) * 16;
    if (m0 >= R || n0 >= Cc) return;            // wave-uniform
    const int mr = lane & 15;
    const int hi = lane >> 4;

    v8f acc = {0.f, 0.f, 0.f, 0.f, 0.f, 0.f, 0.f, 0.f};
    const float* xrow = X + (size_t)(m0 + mr) * ldx;
    const float* wrow = W + (size_t)(n0 + mr) * ldw;

    for (int k0 = 0; k0 < K; k0 += 32) {
        __builtin_prefetch(xrow + k0 + 128, 0, 1);
        v16h a = load_a_tile(xrow, k0, hi);
        // B: lane holds column n = n0+mr, k = hi*16 + e -> 16 consecutive floats
        const float4 b0 = *reinterpret_cast<const float4*>(wrow + k0 + hi * 16);
        const float4 b1 = *reinterpret_cast<const float4*>(wrow + k0 + hi * 16 + 4);
        const float4 b2 = *reinterpret_cast<const float4*>(wrow + k0 + hi * 16 + 8);
        const float4 b3 = *reinterpret_cast<const float4*>(wrow + k0 + hi * 16 + 12);
        v16h b;
        b[0] = (_Float16)b0.x;  b[1] = (_Float16)b0.y;  b[2]  = (_Float16)b0.z;  b[3]  = (_Float16)b0.w;
        b[4] = (_Float16)b1.x;  b[5] = (_Float16)b1.y;  b[6]  = (_Float16)b1.z;  b[7]  = (_Float16)b1.w;
        b[8] = (_Float16)b2.x;  b[9] = (_Float16)b2.y;  b[10] = (_Float16)b2.z;  b[11] = (_Float16)b2.w;
        b[12] = (_Float16)b3.x; b[13] = (_Float16)b3.y; b[14] = (_Float16)b3.z;  b[15] = (_Float16)b3.w;
        acc = __builtin_amdgcn_wmma_f32_16x16x32_f16(false, a, false, b,
                                                     (short)0, acc, false, false);
    }
    gemm_epilogue(acc, Y, ldy, m0, n0, mr, hi, nullptr, alpha, 0, 0);
}

// -------------------------- small helper kernels ---------------------------

__device__ __forceinline__ unsigned fenc(float f) {
    unsigned u = __float_as_uint(f);
    return (u & 0x80000000u) ? ~u : (u | 0x80000000u);
}
__device__ __forceinline__ float fdec(unsigned u) {
    u = (u & 0x80000000u) ? (u & 0x7fffffffu) : ~u;
    return __uint_as_float(u);
}

__global__ __launch_bounds__(CH)
void rowdot2_kernel(const float* __restrict__ h, const float* __restrict__ va,
                    const float* __restrict__ vb, float* __restrict__ da,
                    float* __restrict__ db)
{
    __shared__ float s1[CH], s2[CH];
    int n = blockIdx.x, t = threadIdx.x;
    float x = h[(size_t)n * CH + t];
    s1[t] = x * va[t]; s2[t] = x * vb[t];
    __syncthreads();
    for (int s = CH / 2; s > 0; s >>= 1) {
        if (t < s) { s1[t] += s1[t + s]; s2[t] += s2[t + s]; }
        __syncthreads();
    }
    if (t == 0) { da[n] = s1[0]; db[n] = s2[0]; }
}

__global__ __launch_bounds__(CH)
void gate_kernel(const float* __restrict__ x, const float* __restrict__ w,
                 const float* __restrict__ b, float* __restrict__ gate)
{
    __shared__ float s1[CH];
    int n = blockIdx.x, t = threadIdx.x;
    s1[t] = x[(size_t)n * CH + t] * w[t];
    __syncthreads();
    for (int s = CH / 2; s > 0; s >>= 1) {
        if (t < s) s1[t] += s1[t + s];
        __syncthreads();
    }
    if (t == 0) gate[n] = 1.f / (1.f + __expf(-(s1[0] + b[0])));
}

__global__ __launch_bounds__(256)
void gat_init_kernel(unsigned* __restrict__ menc, float* __restrict__ den, int N)
{
    int i = blockIdx.x * 256 + threadIdx.x;
    if (i < N) { menc[i] = 0x007FFFFFu /* enc(-inf) */; den[i] = 0.f; }
}

__global__ __launch_bounds__(CH)
void init_rows_bias_kernel(float* __restrict__ out, const float* __restrict__ b)
{
    out[(size_t)blockIdx.x * CH + threadIdx.x] = b[threadIdx.x];
}

__global__ __launch_bounds__(256)
void edge_pass1_kernel(const int* __restrict__ ei, int E, int N,
                       const float* __restrict__ hs, const float* __restrict__ hd,
                       float* __restrict__ ebuf, unsigned* __restrict__ menc)
{
    int t = blockIdx.x * 256 + threadIdx.x;
    int ET = E + N;
    if (t >= ET) return;
    int s = (t < E) ? ei[t]     : (t - E);
    int d = (t < E) ? ei[E + t] : (t - E);
    float e = hs[s] + hd[d];
    e = (e > 0.f) ? e : 0.2f * e;               // leaky_relu(0.2)
    ebuf[t] = e;
    atomicMax(&menc[d], fenc(e));
}

__global__ __launch_bounds__(256)
void edge_pass2_kernel(const int* __restrict__ ei, int E, int N,
                       float* __restrict__ ebuf, const unsigned* __restrict__ menc,
                       float* __restrict__ den)
{
    int t = blockIdx.x * 256 + threadIdx.x;
    int ET = E + N;
    if (t >= ET) return;
    int d = (t < E) ? ei[E + t] : (t - E);
    float ex = __expf(ebuf[t] - fdec(menc[d]));
    ebuf[t] = ex;
    atomicAdd(&den[d], ex);
}

__global__ __launch_bounds__(256)
void edge_pass3_kernel(const int* __restrict__ ei, int E, int N,
                       const float* __restrict__ ebuf, const float* __restrict__ den,
                       const float* __restrict__ h, float* __restrict__ out)
{
    int wave = threadIdx.x >> 5, lane = threadIdx.x & 31;
    int t = blockIdx.x * 8 + wave;
    int ET = E + N;
    if (t >= ET) return;
    int s = (t < E) ? ei[t]     : (t - E);
    int d = (t < E) ? ei[E + t] : (t - E);
    float coef = ebuf[t] / den[d];
#pragma unroll
    for (int k = 0; k < 8; ++k) {
        int c = k * 32 + lane;
        atomicAdd(&out[(size_t)d * CH + c], coef * h[(size_t)s * CH + c]);
    }
}

__global__ __launch_bounds__(256)
void softmax_rows_kernel(float* __restrict__ S, int cols)
{
    __shared__ float buf[256];
    int t = threadIdx.x;
    float* p = S + (size_t)blockIdx.x * cols;
    float m = -__builtin_inff();
    for (int c = t; c < cols; c += 256) m = fmaxf(m, p[c]);
    buf[t] = m; __syncthreads();
    for (int s = 128; s > 0; s >>= 1) {
        if (t < s) buf[t] = fmaxf(buf[t], buf[t + s]);
        __syncthreads();
    }
    m = buf[0]; __syncthreads();
    float sum = 0.f;
    for (int c = t; c < cols; c += 256) {
        float ex = __expf(p[c] - m);
        p[c] = ex; sum += ex;
    }
    buf[t] = sum; __syncthreads();
    for (int s = 128; s > 0; s >>= 1) {
        if (t < s) buf[t] += buf[t + s];
        __syncthreads();
    }
    float inv = 1.f / buf[0];
    for (int c = t; c < cols; c += 256) p[c] *= inv;
}

__global__ __launch_bounds__(CH)
void combine_attn_kernel(const float* __restrict__ tA, const float* __restrict__ tB,
                         const float* __restrict__ c1, const float* __restrict__ c2,
                         float* __restrict__ out)
{
    size_t i = (size_t)blockIdx.x * CH + threadIdx.x;
    float s1 = 1.f / (1.f + __expf(-c1[i]));
    float s2 = 1.f / (1.f + __expf(-c2[i]));
    float w1 = 1.f / (1.f + __expf(s2 - s1));
    out[i] = w1 * tA[i] + (1.f - w1) * tB[i];
}

__global__ __launch_bounds__(CH)
void sgu_out_kernel(const float* __restrict__ xin, const float* __restrict__ g,
                    const float* __restrict__ u, const float* __restrict__ alpha,
                    float* __restrict__ out)
{
    size_t i = (size_t)blockIdx.x * CH + threadIdx.x;
    out[i] = xin[i] + alpha[0] * g[i] * u[i];
}

__global__ __launch_bounds__(CH)
void ln_rows_kernel(float* __restrict__ x, const float* __restrict__ g,
                    const float* __restrict__ b)
{
    __shared__ float s1[CH], s2[CH];
    int n = blockIdx.x, t = threadIdx.x;
    float v = x[(size_t)n * CH + t];
    s1[t] = v; s2[t] = v * v;
    __syncthreads();
    for (int s = CH / 2; s > 0; s >>= 1) {
        if (t < s) { s1[t] += s1[t + s]; s2[t] += s2[t + s]; }
        __syncthreads();
    }
    float mean = s1[0] * (1.f / CH);
    float var  = s2[0] * (1.f / CH) - mean * mean;
    float inv  = rsqrtf(var + 1e-5f);
    x[(size_t)n * CH + t] = (v - mean) * inv * g[t] + b[t];
}

__global__ __launch_bounds__(CH)
void msf_combine_kernel(const float* __restrict__ p0, const float* __restrict__ p1,
                        const float* __restrict__ p2, const float* __restrict__ gates,
                        int N, float* __restrict__ out)
{
    int n = blockIdx.x, c = threadIdx.x;
    float g0 = gates[n], g1 = gates[N + n], g2 = gates[2 * N + n];
    float m  = fmaxf(g0, fmaxf(g1, g2));
    float e0 = __expf(g0 - m), e1 = __expf(g1 - m), e2 = __expf(g2 - m);
    float inv = 1.f / (e0 + e1 + e2);
    size_t i = (size_t)n * CH + c;
    out[i] = (p0[i] * e0 + p1[i] * e1 + p2[i] * e2) * inv;
}

// ------------------------------- driver ------------------------------------

extern "C" void kernel_launch(void* const* d_in, const int* in_sizes, int n_in,
                              void* d_out, int out_size, void* d_ws, size_t ws_size,
                              hipStream_t stream)
{
    const int N  = in_sizes[0] / CH;          // 4096
    const int E  = in_sizes[76] / 2;          // 65536
    const int ET = E + N;
    const size_t NC = (size_t)N * CH;
    const int RC = 256;                        // attention row chunk

    // --- input index map (setup_inputs() dict insertion order) ---
    // 0 omics; mf[i] base 1+14i: gatW,a_src,a_dst,gat_b, q_w,q_b,k1_w,k1_b,k2_w,k2_b,v_w,v_b,c1_w,c1_b
    // sgu[i] base 43+5i: up_w,up_b,g_w,g_b,alpha ; msf[i] base 58+6i: p_w,p_b,ln_g,ln_b,w_w,w_b
    // 76 sp_net, 77 om1_net, 78 om2_net
    auto F = [&](int i) { return (const float*)d_in[i]; };
    const float* omics = F(0);
    const int* nets[3] = { (const int*)d_in[76], (const int*)d_in[77], (const int*)d_in[78] };

    // --- workspace carve-up (f32) ---
    float* ws = (float*)d_ws;
    float* h     = ws;            ws += NC;
    float* sp_e  = ws;            ws += NC;
    float* om1_e = ws;            ws += NC;
    float* om2_e = ws;            ws += NC;
    float* emb[3]; for (int i = 0; i < 3; ++i) { emb[i] = ws; ws += NC; }
    float* qb    = ws;            ws += NC;
    float* k1b   = ws;            ws += NC;
    float* k2b   = ws;            ws += NC;
    float* vb    = ws;            ws += NC;
    float* tA    = ws;            ws += NC;
    float* tB    = ws;            ws += NC;
    float* c1b   = ws;            ws += NC;
    float* c2b   = ws;            ws += NC;
    float* ubuf  = ws;            ws += NC;
    float* gbuf  = ws;            ws += NC;
    float* proj[3]; for (int i = 0; i < 3; ++i) { proj[i] = ws; ws += NC; }
    float* Sb    = ws;            ws += (size_t)RC * N;   // 256x4096 score chunk
    float* hs    = ws;            ws += N;
    float* hd    = ws;            ws += N;
    float* den   = ws;            ws += N;
    float* gates = ws;            ws += 3 * (size_t)N;
    float* ebuf  = ws;            ws += ET;
    unsigned* menc = (unsigned*)ws;

    float* outEmb = (float*)d_out;
    float* ups[3] = { outEmb + NC, outEmb + 2 * NC, outEmb + 3 * NC };

    auto gemm = [&](const float* X, int ldx, const float* W, int ldw,
                    const float* bias, float* Y, int ldy,
                    int R, int K, int Cc, float alpha, int accum, int act) {
        dim3 grid(Cc / 16, R / 128), block(256);
        gemm_nn<<<grid, block, 0, stream>>>(X, ldx, W, ldw, bias,
                                            Y, ldy, R, K, Cc, alpha, accum, act);
    };
    auto gemmT = [&](const float* X, int ldx, const float* W, int ldw,
                     float* Y, int ldy, int R, int K, int Cc, float alpha) {
        dim3 grid(Cc / 16, (R + 127) / 128), block(256);
        gemm_nt<<<grid, block, 0, stream>>>(X, ldx, W, ldw, Y, ldy, R, K, Cc, alpha);
    };

    auto run_gat = [&](const float* x, const int* ei, int pb, float* out) {
        gemm(x, CH, F(pb), CH, nullptr, h, CH, N, CH, CH, 1.f, 0, 0);      // h = x@W
        rowdot2_kernel<<<N, CH, 0, stream>>>(h, F(pb + 1), F(pb + 2), hs, hd);
        gat_init_kernel<<<(N + 255) / 256, 256, 0, stream>>>(menc, den, N);
        init_rows_bias_kernel<<<N, CH, 0, stream>>>(out, F(pb + 3));
        edge_pass1_kernel<<<(ET + 255) / 256, 256, 0, stream>>>(ei, E, N, hs, hd, ebuf, menc);
        edge_pass2_kernel<<<(ET + 255) / 256, 256, 0, stream>>>(ei, E, N, ebuf, menc, den);
        edge_pass3_kernel<<<(ET + 7) / 8, 256, 0, stream>>>(ei, E, N, ebuf, den, h, out);
    };

    auto run_attn = [&](const float* sp, const float* o1, const float* o2,
                        int pb, float* out) {
        const float* v_w  = F(pb + 10);
        const float* c1_w = F(pb + 12);
        gemm(sp, CH, F(pb + 4), CH, F(pb + 5), qb,  CH, N, CH, CH, 1.f, 0, 0); // q
        gemm(o1, CH, F(pb + 6), CH, F(pb + 7), k1b, CH, N, CH, CH, 1.f, 0, 0); // k1
        gemm(o2, CH, F(pb + 8), CH, F(pb + 9), k2b, CH, N, CH, CH, 1.f, 0, 0); // k2
        // v = cat(sp,o1,o2) @ v_w + v_b  (three accumulating GEMMs on row blocks)
        gemm(sp, CH, v_w,               CH, F(pb + 11), vb, CH, N, CH, CH, 1.f, 0, 0);
        gemm(o1, CH, v_w + CH * CH,     CH, nullptr,    vb, CH, N, CH, CH, 1.f, 1, 0);
        gemm(o2, CH, v_w + 2 * CH * CH, CH, nullptr,    vb, CH, N, CH, CH, 1.f, 1, 0);
        // dense attentions, chunked over query rows to bound scratch
        const float scale = 0.0625f;  // 1/sqrt(256)
        for (int a = 0; a < 2; ++a) {
            const float* kb = a ? k2b : k1b;
            float* tdst = a ? tB : tA;
            for (int r0 = 0; r0 < N; r0 += RC) {
                gemmT(qb + (size_t)r0 * CH, CH, kb, CH, Sb, N, RC, CH, N, scale); // S=q@k^T*s
                softmax_rows_kernel<<<RC, 256, 0, stream>>>(Sb, N);
                gemm(Sb, N, vb, CH, nullptr,
                     tdst + (size_t)r0 * CH, CH, RC, N, CH, 1.f, 0, 0);           // P@v
            }
        }
        // confidences (conf2 reuses c1 weights, faithful to the reference)
        gemm(sp, CH, c1_w,           CH, F(pb + 13), c1b, CH, N, CH, CH, 1.f, 0, 0);
        gemm(o1, CH, c1_w + CH * CH, CH, nullptr,    c1b, CH, N, CH, CH, 1.f, 1, 0);
        gemm(sp, CH, c1_w,           CH, F(pb + 13), c2b, CH, N, CH, CH, 1.f, 0, 0);
        gemm(o2, CH, c1_w + CH * CH, CH, nullptr,    c2b, CH, N, CH, CH, 1.f, 1, 0);
        combine_attn_kernel<<<N, CH, 0, stream>>>(tA, tB, c1b, c2b, out);
    };

    auto run_sgu = [&](const float* xin, const float* xup, int pb, float* out) {
        gemm(xup, CH, F(pb), CH, F(pb + 1), ubuf, CH, N, CH, CH, 1.f, 0, 2);     // u = tanh
        gemm(xin,  CH, F(pb + 2),           CH, F(pb + 3), gbuf, CH, N, CH, CH, 1.f, 0, 0);
        gemm(ubuf, CH, F(pb + 2) + CH * CH, CH, nullptr,   gbuf, CH, N, CH, CH, 1.f, 1, 1); // sigmoid
        sgu_out_kernel<<<N, CH, 0, stream>>>(xin, gbuf, ubuf, F(pb + 4), out);
    };

    // ---------------- encoder pipeline ----------------
    run_gat(omics, nets[0], 1, sp_e);
    run_gat(omics, nets[1], 1, om1_e);
    run_gat(omics, nets[2], 1, om2_e);
    run_attn(sp_e, om1_e, om2_e, 1, emb[0]);
    for (int i = 1; i < 3; ++i) {
        int pb = 1 + 14 * i;
        run_gat(emb[i - 1], nets[0], pb, sp_e);
        run_attn(sp_e, om1_e, om2_e, pb, emb[i]);
    }
    run_sgu(emb[0], emb[0], 43, ups[0]);
    run_sgu(emb[1], emb[0], 48, ups[1]);
    run_sgu(emb[2], emb[1], 53, ups[2]);
    for (int i = 0; i < 3; ++i) {
        int pb = 58 + 6 * i;
        gemm(ups[i], CH, F(pb), CH, F(pb + 1), proj[i], CH, N, CH, CH, 1.f, 0, 0);
        ln_rows_kernel<<<N, CH, 0, stream>>>(proj[i], F(pb + 2), F(pb + 3));
        gate_kernel<<<N, CH, 0, stream>>>(proj[i], F(pb + 4), F(pb + 5), gates + (size_t)i * N);
    }
    msf_combine_kernel<<<N, CH, 0, stream>>>(proj[0], proj[1], proj[2], gates, N, outEmb);
}
